// SocialPlannerAgent_50998441673265
// MI455X (gfx1250) — matrix operand order repaired and checked by reading, not compiled, not run on hardware
//
#include <hip/hip_runtime.h>
#include <hip/hip_bf16.h>
#include <math.h>

// =====================================================================
// GraphNet (SocialPlannerAgent) for MI455X / gfx1250, wave32 + WMMA.
// Heavy path (edge MLP, ~26 GFLOP) on v_wmma_f32_16x16x32_f16 with f32
// accumulate. Weights pre-packed into per-lane WMMA-B fragment order so
// each B fragment is one aligned 32B load. Waves process M=32 rows
// (two A fragments, 8 f32 accumulators) so every B fragment feeds two
// WMMAs. h_e (64MB) stays L2-resident (192MB L2); in block 2 the policy
// head is fused into the edge kernel so final h_e is never written back
// or re-read (saves 128MB of L2 traffic + a launch).
// =====================================================================

typedef __attribute__((ext_vector_type(16))) _Float16 v16h;
typedef __attribute__((ext_vector_type(8)))  _Float16 v8h;
typedef __attribute__((ext_vector_type(4)))  _Float16 v4h;
typedef __attribute__((ext_vector_type(2)))  _Float16 v2h;
typedef __attribute__((ext_vector_type(8)))  float    v8f;

#define HID 64
#define NN  128
#define BB  16
#define NEDGE (BB * NN * NN)          // 262144
#define NNODE (BB * NN)               // 2048
#define EDGE_TILES32 (NEDGE / 32)     // 8192
#define NODE_TILES32 (NNODE / 32)     // 64
#define HSTRIDE 80                    // padded f16 row stride in LDS (160B)

// ---------------- workspace layout (float offsets) -------------------
#define OFF_HX   0                         // 131072 f32
#define OFF_HU   131072                    // 1024 f32
#define OFF_INC  132096                    // 131072 f32 (incoming)
#define OFF_NAGG 263168                    // 1024 f32  (node_agg) contiguous with INC
#define OFF_EAGG 264192                    // 1024 f32  (edge_agg)
#define OFF_W16  265216                    // f16 packed-weight region (139264 halves)
#define OFF_HE   334848                    // 16777216 f32 (h_e), 32B aligned
// f16 region offsets (halves), per graph-block stride 65536:
#define W16_EW1 0
#define W16_EW2 16384
#define W16_EW3 20480
#define W16_NW1 24576
#define W16_NW2 36864
#define W16_NW3 40960
#define W16_GW1 45056
#define W16_GW2 57344
#define W16_GW3 61440
#define W16_BLK 65536
#define W16_PW  131072
#define W16_VW  135168

// ---------------- fragment helpers -----------------------------------
// A fragment (16x32 f16, MxK): lane l -> row ro+(l&15), kb=8*(l>=16);
// halves are two contiguous runs: [k0+kb, +8) and [k0+kb+16, +8).
// rp(m,c) returns a pointer to 8 contiguous f32 features c..c+7 of row m
// (runs never cross a 64-wide concat segment: c mod 64 in {0,8,...,56}).
template <typename RF>
__device__ __forceinline__ v16h make_A_rt(RF rp, int ro, int lane, int k0) {
  const int m  = ro + (lane & 15);
  const int kb = (lane & 16) ? 8 : 0;
  const float* p0 = rp(m, k0 + kb);
  const float* p1 = rp(m, k0 + kb + 16);
  const float4 q0 = *(const float4*)(p0);
  const float4 q1 = *(const float4*)(p0 + 4);
  const float4 q2 = *(const float4*)(p1);
  const float4 q3 = *(const float4*)(p1 + 4);
  v16h a;
  a[0]  = (_Float16)q0.x; a[1]  = (_Float16)q0.y; a[2]  = (_Float16)q0.z; a[3]  = (_Float16)q0.w;
  a[4]  = (_Float16)q1.x; a[5]  = (_Float16)q1.y; a[6]  = (_Float16)q1.z; a[7]  = (_Float16)q1.w;
  a[8]  = (_Float16)q2.x; a[9]  = (_Float16)q2.y; a[10] = (_Float16)q2.z; a[11] = (_Float16)q2.w;
  a[12] = (_Float16)q3.x; a[13] = (_Float16)q3.y; a[14] = (_Float16)q3.z; a[15] = (_Float16)q3.w;
  return a;
}

// A fragment from f16 LDS buffer (row stride HSTRIDE): two aligned b128 loads.
__device__ __forceinline__ v16h load_A_lds(const _Float16* h, int ro, int lane, int k0) {
  const int m  = ro + (lane & 15);
  const int kb = (lane & 16) ? 8 : 0;
  const v8h lo = *(const v8h*)(h + m * HSTRIDE + k0 + kb);
  const v8h hi = *(const v8h*)(h + m * HSTRIDE + k0 + kb + 16);
  v16h a;
#pragma unroll
  for (int e = 0; e < 8; ++e) { a[e] = lo[e]; a[8 + e] = hi[e]; }
  return a;
}

#define WMMA_F16(A, B, C) \
  __builtin_amdgcn_wmma_f32_16x16x32_f16(false, (A), false, (B), (short)0, (C), false, false)

// 32xK @ Kx64 -> f32 act tile (32x64 LDS). Wp is packed fragment-major:
// frag = (k0/32)*4 + nt, layout [frag][lane][16 halves] (one 32B load).
// Each B fragment feeds two WMMAs (rows 0-15 and 16-31).
// C/D layout: lane -> col n=l&15, vgpr v -> row m = v + 8*(l>=16).
template <int K, typename FA>
__device__ __forceinline__ void gemm32xK(FA fa, const _Float16* __restrict__ Wp,
                                         const float* __restrict__ bias, bool relu,
                                         float* act, int lane) {
  v8f acc0[4], acc1[4];
#pragma unroll
  for (int nt = 0; nt < 4; ++nt) { acc0[nt] = (v8f){}; acc1[nt] = (v8f){}; }
#pragma unroll
  for (int k0 = 0; k0 < K; k0 += 32) {
    const v16h a0 = fa(0, lane, k0);
    const v16h a1 = fa(16, lane, k0);
    const _Float16* wp = Wp + (k0 >> 5) * 2048 + lane * 16;
#pragma unroll
    for (int nt = 0; nt < 4; ++nt) {
      const v16h bfr = *(const v16h*)(wp + nt * 512);
      acc0[nt] = WMMA_F16(a0, bfr, acc0[nt]);
      acc1[nt] = WMMA_F16(a1, bfr, acc1[nt]);
    }
  }
  const int nl = lane & 15;
  const int mo = (lane & 16) ? 8 : 0;
#pragma unroll
  for (int nt = 0; nt < 4; ++nt) {
    const int n = nt * 16 + nl;
    const float bv = bias[n];
#pragma unroll
    for (int v = 0; v < 8; ++v) {
      float x0 = acc0[nt][v] + bv;
      float x1 = acc1[nt][v] + bv;
      if (relu) { x0 = fmaxf(x0, 0.0f); x1 = fmaxf(x1, 0.0f); }
      act[(mo + v) * HID + n]      = x0;
      act[(16 + mo + v) * HID + n] = x1;
    }
  }
}

// 16xK variant for the tiny global/value paths.
template <int K, typename FA>
__device__ __forceinline__ void gemm16xK(FA fa, const _Float16* __restrict__ Wp,
                                         const float* __restrict__ bias, bool relu,
                                         float* act, int lane) {
  v8f acc[4];
#pragma unroll
  for (int nt = 0; nt < 4; ++nt) acc[nt] = (v8f){};
#pragma unroll
  for (int k0 = 0; k0 < K; k0 += 32) {
    const v16h a = fa(0, lane, k0);
    const _Float16* wp = Wp + (k0 >> 5) * 2048 + lane * 16;
#pragma unroll
    for (int nt = 0; nt < 4; ++nt) {
      const v16h bfr = *(const v16h*)(wp + nt * 512);
      acc[nt] = WMMA_F16(a, bfr, acc[nt]);
    }
  }
  const int nl = lane & 15;
  const int mo = (lane & 16) ? 8 : 0;
#pragma unroll
  for (int nt = 0; nt < 4; ++nt) {
    const int n = nt * 16 + nl;
    const float bv = bias[n];
#pragma unroll
    for (int v = 0; v < 8; ++v) {
      float xv = acc[nt][v] + bv;
      if (relu) xv = fmaxf(xv, 0.0f);
      act[(mo + v) * HID + n] = xv;
    }
  }
}

// LayerNorm, 32 rows: one row per lane; float4 LDS reads, v4h (8B) stores.
__device__ __forceinline__ void wave_ln32(const float* act, const float* __restrict__ g,
                                          const float* __restrict__ bp, _Float16* out, int lane) {
  const float* ar = act + lane * HID;
  float s = 0.f, ss = 0.f;
#pragma unroll
  for (int q = 0; q < 16; ++q) {
    const float4 v = *(const float4*)(ar + 4 * q);
    s  += v.x + v.y + v.z + v.w;
    ss += v.x * v.x + v.y * v.y + v.z * v.z + v.w * v.w;
  }
  const float mean = s * 0.015625f;
  const float var  = ss * 0.015625f - mean * mean;
  const float rstd = rsqrtf(var + 1e-5f);
#pragma unroll
  for (int q = 0; q < 16; ++q) {
    const float4 v  = *(const float4*)(ar + 4 * q);
    const float4 g4 = *(const float4*)(g + 4 * q);
    const float4 b4 = *(const float4*)(bp + 4 * q);
    v4h o;
    o[0] = (_Float16)(g4.x * ((v.x - mean) * rstd) + b4.x);
    o[1] = (_Float16)(g4.y * ((v.y - mean) * rstd) + b4.y);
    o[2] = (_Float16)(g4.z * ((v.z - mean) * rstd) + b4.z);
    o[3] = (_Float16)(g4.w * ((v.w - mean) * rstd) + b4.w);
    *(v4h*)(out + lane * HSTRIDE + 4 * q) = o;
  }
}

// LayerNorm, 16 rows: 2 lanes per row + shfl_xor(16) (global/value paths).
__device__ __forceinline__ void wave_ln16(const float* act, const float* __restrict__ g,
                                          const float* __restrict__ bp, _Float16* out, int lane) {
  const int row = lane & 15;
  const int hf  = lane >> 4;
  float s = 0.f, ss = 0.f;
#pragma unroll
  for (int k = 0; k < 32; ++k) {
    const float v = act[row * HID + hf * 32 + k];
    s += v; ss += v * v;
  }
  s  += __shfl_xor(s, 16, 32);
  ss += __shfl_xor(ss, 16, 32);
  const float mean = s * 0.015625f;
  const float var  = ss * 0.015625f - mean * mean;
  const float rstd = rsqrtf(var + 1e-5f);
#pragma unroll
  for (int k = 0; k < 16; ++k) {
    const int c = hf * 32 + 2 * k;
    const float v0 = (act[row * HID + c] - mean) * rstd;
    const float v1 = (act[row * HID + c + 1] - mean) * rstd;
    v2h o;
    o[0] = (_Float16)(g[c] * v0 + bp[c]);
    o[1] = (_Float16)(g[c + 1] * v1 + bp[c + 1]);
    *(v2h*)(out + row * HSTRIDE + c) = o;
  }
}

// LN1 -> GEMM2(relu) -> LN2 -> GEMM3, 32-row tiles.
__device__ __forceinline__ void mlp_tail32(float* act, _Float16* hbuf,
    const float* g1, const float* be1,
    const _Float16* W2, const float* b2, const float* g2, const float* be2,
    const _Float16* W3, const float* b3, int lane) {
  wave_ln32(act, g1, be1, hbuf, lane);
  __builtin_amdgcn_wave_barrier();
  gemm32xK<64>([&](int ro, int ln, int k0) { return load_A_lds(hbuf, ro, ln, k0); },
               W2, b2, true, act, lane);
  __builtin_amdgcn_wave_barrier();
  wave_ln32(act, g2, be2, hbuf, lane);
  __builtin_amdgcn_wave_barrier();
  gemm32xK<64>([&](int ro, int ln, int k0) { return load_A_lds(hbuf, ro, ln, k0); },
               W3, b3, false, act, lane);
  __builtin_amdgcn_wave_barrier();
}

__device__ __forceinline__ void mlp_tail16(float* act, _Float16* hbuf,
    const float* g1, const float* be1,
    const _Float16* W2, const float* b2, const float* g2, const float* be2,
    const _Float16* W3, const float* b3, int lane) {
  wave_ln16(act, g1, be1, hbuf, lane);
  __builtin_amdgcn_wave_barrier();
  gemm16xK<64>([&](int ro, int ln, int k0) { return load_A_lds(hbuf, ro, ln, k0); },
               W2, b2, true, act, lane);
  __builtin_amdgcn_wave_barrier();
  wave_ln16(act, g2, be2, hbuf, lane);
  __builtin_amdgcn_wave_barrier();
  gemm16xK<64>([&](int ro, int ln, int k0) { return load_A_lds(hbuf, ro, ln, k0); },
               W3, b3, false, act, lane);
  __builtin_amdgcn_wave_barrier();
}

// ---------------- kernels --------------------------------------------
// Pack a K x 64 f32 weight into WMMA-B fragment-major f16 layout:
// dst[frag*512 + lane*16 + e] = W[(kc*32 + 8*(lane>=16) + e + 8*(e>=8))*64 + nt*16 + (lane&15)]
__global__ void pack_B_kernel(const float* __restrict__ W, _Float16* __restrict__ dst, int K) {
  const int t = blockIdx.x * 128 + threadIdx.x;   // frag*32 + lane
  const int nfrag = (K >> 5) * 4;
  if (t >= nfrag * 32) return;
  const int frag = t >> 5, lane = t & 31;
  const int kc = frag >> 2, nt = frag & 3;
  const int k0 = kc * 32 + ((lane & 16) ? 8 : 0);
  const int n  = nt * 16 + (lane & 15);
  _Float16* o = dst + frag * 512 + lane * 16;
#pragma unroll
  for (int e = 0; e < 16; ++e) o[e] = (_Float16)W[(k0 + e + ((e >= 8) ? 8 : 0)) * HID + n];
}

__global__ void zero_kernel(float* __restrict__ p, int n) {
  const int i = blockIdx.x * 256 + threadIdx.x;
  if (i < n) p[i] = 0.0f;
}

__global__ void enc_small_kernel(const float* __restrict__ x, const float* __restrict__ u,
                                 const float* __restrict__ Wn, const float* __restrict__ bn,
                                 const float* __restrict__ Wg, const float* __restrict__ bg,
                                 float* __restrict__ hx, float* __restrict__ hu) {
  const int idx = blockIdx.x * 256 + threadIdx.x;
  if (idx < NNODE * HID) {
    const int row = idx >> 6, k = idx & 63;
    const float v = x[row * 2] * Wn[k] + x[row * 2 + 1] * Wn[HID + k] + bn[k];
    hx[idx] = fmaxf(v, 0.0f);
  } else if (idx < NNODE * HID + BB * HID) {
    const int t = idx - NNODE * HID;
    const int b = t >> 6, k = t & 63;
    hu[t] = fmaxf(u[b] * Wg[k] + bg[k], 0.0f);
  }
}

__global__ void enc_edge_kernel(const float* __restrict__ ea, const float* __restrict__ We,
                                const float* __restrict__ be, float* __restrict__ he) {
  const int idx = blockIdx.x * 256 + threadIdx.x;   // one float4 per thread
  if (idx >= NEDGE * 16) return;                    // 64 feats = 16 float4 per edge
  const int e  = idx >> 4;
  const int q  = idx & 15;
  const float ev = ea[e];
  const float4 w   = ((const float4*)We)[q];
  const float4 bb4 = ((const float4*)be)[q];
  float4 o;
  o.x = fmaxf(ev * w.x + bb4.x, 0.0f);
  o.y = fmaxf(ev * w.y + bb4.y, 0.0f);
  o.z = fmaxf(ev * w.z + bb4.z, 0.0f);
  o.w = fmaxf(ev * w.w + bb4.w, 0.0f);
  ((float4*)he)[idx] = o;
}

// Edge MLP block. FUSE_POLICY=false: write h_e += new_edge (residual).
// FUSE_POLICY=true (last block): h_e is dead after the policy head, so
// skip the residual write-back; instead form h_final = h_e + new_edge in
// f16 LDS and run the fused policy head (WMMA 64->64 + dot 64->1) here.
template <bool FUSE_POLICY>
__global__ void __launch_bounds__(128) edge_block_kernel(
    const float* __restrict__ hx, float* __restrict__ he, const float* __restrict__ hu,
    const _Float16* __restrict__ W1, const float* __restrict__ b1,
    const float* __restrict__ g1, const float* __restrict__ be1,
    const _Float16* __restrict__ W2, const float* __restrict__ b2,
    const float* __restrict__ g2, const float* __restrict__ be2,
    const _Float16* __restrict__ W3, const float* __restrict__ b3,
    float* __restrict__ incoming,
    const _Float16* __restrict__ Wp1, const float* __restrict__ bp1,
    const float* __restrict__ wp2, const float* __restrict__ bp2,
    float* __restrict__ out, int tilesPerWave) {
  __shared__ __align__(32) float    act_s[4][32 * HID];
  __shared__ __align__(32) _Float16 h_s[4][32 * HSTRIDE];
  const int lane = threadIdx.x & 31;
  const int wv   = threadIdx.x >> 5;
  float*    act  = act_s[wv];
  _Float16* hbuf = h_s[wv];
  const int waveGlobal = blockIdx.x * 4 + wv;
  for (int t = 0; t < tilesPerWave; ++t) {
    const int tile = waveGlobal * tilesPerWave + t;
    if (tile >= EDGE_TILES32) return;
    const int e0 = tile << 5;           // 32 consecutive edges, same (b,i)
    const int b  = e0 >> 14;
    const int i  = (e0 >> 7) & 127;
    const int j0 = e0 & 127;
    // run pointer into the 256-wide concat [x_i | x_j | h_e | h_u]
    auto runp = [&](int m, int c) -> const float* {
      if (c < 64)  return hx + (b * NN + i) * HID + c;
      if (c < 128) return hx + (b * NN + (j0 + m)) * HID + (c - 64);
      if (c < 192) return he + (e0 + m) * HID + (c - 128);
      return hu + b * HID + (c - 192);
    };
    gemm32xK<256>([&](int ro, int ln, int k0) { return make_A_rt(runp, ro, ln, k0); },
                  W1, b1, true, act, lane);
    __builtin_amdgcn_wave_barrier();
    mlp_tail32(act, hbuf, g1, be1, W2, b2, g2, be2, W3, b3, lane);
    // per-(b,i) aggregation: column sums of new_edge over the 32 rows
    for (int cc = lane; cc < HID; cc += 32) {
      float s = 0.0f;
#pragma unroll
      for (int m = 0; m < 32; ++m) s += act[m * HID + cc];
      atomicAdd(&incoming[(b * NN + i) * HID + cc], s);
    }
    const float* het = he + e0 * HID;   // tile = 2048 contiguous f32 of h_e
    if (!FUSE_POLICY) {
      // residual h_e += new_edge: coalesced float4 RMW
#pragma unroll
      for (int q = 0; q < 16; ++q) {
        const int t4 = q * 32 + lane;
        float4 h4 = *(const float4*)(het + t4 * 4);
        const float4 a4 = *(const float4*)(act + t4 * 4);
        h4.x += a4.x; h4.y += a4.y; h4.z += a4.z; h4.w += a4.w;
        *(float4*)(he + e0 * HID + t4 * 4) = h4;
      }
    } else {
      // h_final = h_e + new_edge -> f16 LDS (no global write-back)
      __builtin_amdgcn_wave_barrier();
#pragma unroll
      for (int q = 0; q < 16; ++q) {
        const int t4 = q * 32 + lane;
        const float4 h4 = *(const float4*)(het + t4 * 4);
        const float4 a4 = *(const float4*)(act + t4 * 4);
        v4h o;
        o[0] = (_Float16)(h4.x + a4.x);
        o[1] = (_Float16)(h4.y + a4.y);
        o[2] = (_Float16)(h4.z + a4.z);
        o[3] = (_Float16)(h4.w + a4.w);
        const int row = t4 >> 4, col4 = (t4 & 15) * 4;
        *(v4h*)(hbuf + row * HSTRIDE + col4) = o;
      }
      __builtin_amdgcn_wave_barrier();
      // fused policy head: relu(h_final @ Wp1 + bp1) @ wp2 + bp2
      gemm32xK<64>([&](int ro, int ln, int k0) { return load_A_lds(hbuf, ro, ln, k0); },
                   Wp1, bp1, true, act, lane);
      __builtin_amdgcn_wave_barrier();
      float s = 0.0f;
#pragma unroll
      for (int q = 0; q < 16; ++q) {
        const float4 a4 = *(const float4*)(act + lane * HID + 4 * q);
        const float4 w4 = *(const float4*)(wp2 + 4 * q);
        s += a4.x * w4.x + a4.y * w4.y + a4.z * w4.z + a4.w * w4.w;
      }
      const int e  = e0 + lane;
      const int ii = (e >> 7) & 127, jj = e & 127;
      out[e] = (ii == jj) ? -__builtin_inff() : (s + bp2[0]);
      __builtin_amdgcn_wave_barrier();
    }
  }
}

__global__ void reduce_edge_agg_kernel(const float* __restrict__ incoming, float* __restrict__ eagg) {
  const int idx = blockIdx.x * 256 + threadIdx.x;   // 1024 outputs
  if (idx >= BB * HID) return;
  const int b = idx >> 6, k = idx & 63;
  float s = 0.0f;
  for (int i = 0; i < NN; ++i) s += incoming[(b * NN + i) * HID + k];
  eagg[idx] = s;
}

__global__ void __launch_bounds__(128) node_block_kernel(
    float* __restrict__ hx, const float* __restrict__ incoming, const float* __restrict__ hu,
    const _Float16* __restrict__ W1, const float* __restrict__ b1,
    const float* __restrict__ g1, const float* __restrict__ be1,
    const _Float16* __restrict__ W2, const float* __restrict__ b2,
    const float* __restrict__ g2, const float* __restrict__ be2,
    const _Float16* __restrict__ W3, const float* __restrict__ b3,
    float* __restrict__ nagg) {
  __shared__ __align__(32) float    act_s[4][32 * HID];
  __shared__ __align__(32) _Float16 h_s[4][32 * HSTRIDE];
  const int lane = threadIdx.x & 31;
  const int wv   = threadIdx.x >> 5;
  float*    act  = act_s[wv];
  _Float16* hbuf = h_s[wv];
  const int tile = blockIdx.x * 4 + wv;
  if (tile >= NODE_TILES32) return;
  const int r0 = tile << 5;             // 32 consecutive nodes, same b
  const int b  = r0 >> 7;
  auto runp = [&](int m, int c) -> const float* {
    if (c < 64)  return hx + (r0 + m) * HID + c;
    if (c < 128) return incoming + (r0 + m) * HID + (c - 64);
    return hu + b * HID + (c - 128);
  };
  gemm32xK<192>([&](int ro, int ln, int k0) { return make_A_rt(runp, ro, ln, k0); },
                W1, b1, true, act, lane);
  __builtin_amdgcn_wave_barrier();
  mlp_tail32(act, hbuf, g1, be1, W2, b2, g2, be2, W3, b3, lane);
  float* hxt = hx + r0 * HID;
#pragma unroll
  for (int q = 0; q < 16; ++q) {
    const int t4 = q * 32 + lane;
    float4 h4 = *(const float4*)(hxt + t4 * 4);
    const float4 a4 = *(const float4*)(act + t4 * 4);
    h4.x += a4.x; h4.y += a4.y; h4.z += a4.z; h4.w += a4.w;
    *(float4*)(hxt + t4 * 4) = h4;
  }
  for (int cc = lane; cc < HID; cc += 32) {
    float s = 0.0f;
#pragma unroll
    for (int m = 0; m < 32; ++m) s += act[m * HID + cc];
    atomicAdd(&nagg[b * HID + cc], s);
  }
}

__global__ void global_block_kernel(
    float* __restrict__ hu, const float* __restrict__ nagg, const float* __restrict__ eagg,
    const _Float16* __restrict__ W1, const float* __restrict__ b1,
    const float* __restrict__ g1, const float* __restrict__ be1,
    const _Float16* __restrict__ W2, const float* __restrict__ b2,
    const float* __restrict__ g2, const float* __restrict__ be2,
    const _Float16* __restrict__ W3, const float* __restrict__ b3) {
  __shared__ __align__(32) float    act[16 * HID];
  __shared__ __align__(32) _Float16 hbuf[16 * HSTRIDE];
  const int lane = threadIdx.x;
  auto runp = [&](int m, int c) -> const float* {
    if (c < 64)  return nagg + m * HID + c;
    if (c < 128) return eagg + m * HID + (c - 64);
    return hu + m * HID + (c - 128);
  };
  gemm16xK<192>([&](int ro, int ln, int k0) { return make_A_rt(runp, ro, ln, k0); },
                W1, b1, true, act, lane);
  __builtin_amdgcn_wave_barrier();
  mlp_tail16(act, hbuf, g1, be1, W2, b2, g2, be2, W3, b3, lane);
  for (int cc = lane; cc < HID; cc += 32)
    for (int m = 0; m < 16; ++m) hu[m * HID + cc] += act[m * HID + cc];
}

__global__ void value_kernel(const float* __restrict__ hu, const _Float16* __restrict__ Wv1,
                             const float* __restrict__ bv1, const float* __restrict__ wv2,
                             const float* __restrict__ bv2, float* __restrict__ out) {
  __shared__ __align__(32) float act[16 * HID];
  const int lane = threadIdx.x;
  auto runp = [&](int m, int c) -> const float* { return hu + m * HID + c; };
  gemm16xK<64>([&](int ro, int ln, int k0) { return make_A_rt(runp, ro, ln, k0); },
               Wv1, bv1, true, act, lane);
  __builtin_amdgcn_wave_barrier();
  const int row = lane & 15, hf = lane >> 4;
  float s = 0.0f;
#pragma unroll
  for (int k = 0; k < 32; ++k) s += act[row * HID + hf * 32 + k] * wv2[hf * 32 + k];
  s += __shfl_xor(s, 16, 32);
  if (lane < 16) out[row] = s + bv2[0];
}

// ---------------- host side ------------------------------------------
// Param leaf order (JAX pytree, dict keys sorted):
// 0:x 1:edge_attr 2:u, then params:
//  blocks[bi] base = 3+30*bi:
//    edge_mlp  +0..9 : l1.W,l1.b,l2.W,l2.b,l3.W,l3.b,ln1.b,ln1.g,ln2.b,ln2.g
//    global_mlp +10..19, node_mlp +20..29 (same inner order)
//  63:edge_enc.W 64:edge_enc.b 65:glob_enc.W 66:glob_enc.b
//  67:node_enc.W 68:node_enc.b
//  69:policy0.W 70:policy0.b 71:policy1.W 72:policy1.b
//  73:value0.W  74:value0.b  75:value1.W  76:value1.b

extern "C" void kernel_launch(void* const* d_in, const int* in_sizes, int n_in,
                              void* d_out, int out_size, void* d_ws, size_t ws_size,
                              hipStream_t stream) {
  (void)in_sizes; (void)n_in; (void)out_size; (void)ws_size;
  auto P = [&](int i) { return (const float*)d_in[i]; };
  float* ws = (float*)d_ws;
  float* hx   = ws + OFF_HX;
  float* hu   = ws + OFF_HU;
  float* inc  = ws + OFF_INC;
  float* nagg = ws + OFF_NAGG;
  float* eagg = ws + OFF_EAGG;
  float* he   = ws + OFF_HE;
  _Float16* Wf = (_Float16*)(ws + OFF_W16);
  float* out = (float*)d_out;

  // 1. pack all matmul weights into f16 fragment-major layout
  auto pack = [&](const float* src, _Float16* dst, int K) {
    const int nthr = K * 4;   // (K/32)*4 frags * 32 lanes
    pack_B_kernel<<<(nthr + 127) / 128, 128, 0, stream>>>(src, dst, K);
  };
  for (int bi = 0; bi < 2; ++bi) {
    const int base = 3 + 30 * bi;
    _Float16* wb = Wf + bi * W16_BLK;
    pack(P(base + 0),  wb + W16_EW1, 256);
    pack(P(base + 2),  wb + W16_EW2, 64);
    pack(P(base + 4),  wb + W16_EW3, 64);
    pack(P(base + 20), wb + W16_NW1, 192);
    pack(P(base + 22), wb + W16_NW2, 64);
    pack(P(base + 24), wb + W16_NW3, 64);
    pack(P(base + 10), wb + W16_GW1, 192);
    pack(P(base + 12), wb + W16_GW2, 64);
    pack(P(base + 14), wb + W16_GW3, 64);
  }
  pack(P(69), Wf + W16_PW, 64);
  pack(P(73), Wf + W16_VW, 64);

  // 2. encoders
  enc_small_kernel<<<(NNODE * HID + BB * HID + 255) / 256, 256, 0, stream>>>(
      P(0), P(2), P(67), P(68), P(65), P(66), hx, hu);
  enc_edge_kernel<<<(NEDGE * 16 + 255) / 256, 256, 0, stream>>>(P(1), P(63), P(64), he);

  // 3. two GraphNet blocks (policy head fused into block-2 edge kernel)
  for (int bi = 0; bi < 2; ++bi) {
    const int base = 3 + 30 * bi;
    _Float16* wb = Wf + bi * W16_BLK;
    zero_kernel<<<(132096 + 255) / 256, 256, 0, stream>>>(inc, 132096);  // incoming + node_agg
    if (bi == 0) {
      edge_block_kernel<false><<<EDGE_TILES32 / 16, 128, 0, stream>>>(
          hx, he, hu,
          wb + W16_EW1, P(base + 1), P(base + 7), P(base + 6),
          wb + W16_EW2, P(base + 3), P(base + 9), P(base + 8),
          wb + W16_EW3, P(base + 5),
          inc, nullptr, nullptr, nullptr, nullptr, nullptr, 4);
    } else {
      edge_block_kernel<true><<<EDGE_TILES32 / 16, 128, 0, stream>>>(
          hx, he, hu,
          wb + W16_EW1, P(base + 1), P(base + 7), P(base + 6),
          wb + W16_EW2, P(base + 3), P(base + 9), P(base + 8),
          wb + W16_EW3, P(base + 5),
          inc, Wf + W16_PW, P(70), P(71), P(72), out, 4);
    }
    reduce_edge_agg_kernel<<<4, 256, 0, stream>>>(inc, eagg);
    node_block_kernel<<<NODE_TILES32 / 4, 128, 0, stream>>>(
        hx, inc, hu,
        wb + W16_NW1, P(base + 21), P(base + 27), P(base + 26),
        wb + W16_NW2, P(base + 23), P(base + 29), P(base + 28),
        wb + W16_NW3, P(base + 25),
        nagg);
    global_block_kernel<<<1, 32, 0, stream>>>(
        hu, nagg, eagg,
        wb + W16_GW1, P(base + 11), P(base + 17), P(base + 16),
        wb + W16_GW2, P(base + 13), P(base + 19), P(base + 18),
        wb + W16_GW3, P(base + 15));
  }

  // 4. value head
  value_kernel<<<1, 32, 0, stream>>>(hu, Wf + W16_VW, P(74), P(75), P(76), out + NEDGE);
}
